// MultiHeadSpatialAttention_43215960933223
// MI455X (gfx1250) — matrix operand order repaired and checked by reading, not compiled, not run on hardware
//
#include <hip/hip_runtime.h>
#include <hip/hip_bf16.h>
#include <math.h>

// ---------------------------------------------------------------------------
// MultiHeadSpatialAttention for MI455X (gfx1250, wave32, WMMA bf16 f32-acc)
// B=8, C=512, T=32*32=1024, HEADS=8, HD=64, GROUPS=32
// Data path: global -> GLOBAL_LOAD_ASYNC_TO_LDS_B128 (ASYNCcnt) -> LDS ->
//            contiguous ds_load_b128 fragments -> v_wmma_f32_16x16x32_bf16
// All B-operand tensors stored k-contiguous (column-major) so each lane's
// fragment is one contiguous 32-byte run.
// ---------------------------------------------------------------------------

typedef __attribute__((ext_vector_type(16))) __bf16 v16bf;
typedef __attribute__((ext_vector_type(8)))  float  v8f;
typedef __attribute__((ext_vector_type(4)))  int    v4i;

typedef __attribute__((address_space(1))) v4i gv4i_t;  // global int4
typedef __attribute__((address_space(3))) v4i lv4i_t;  // LDS int4

#define DEV __device__ __forceinline__

DEV __bf16 f2bf(float f) {
  unsigned u = __float_as_uint(f);
  u = (u + 0x7FFFu + ((u >> 16) & 1u)) >> 16;       // round-to-nearest-even
  unsigned short s = (unsigned short)u;
  return __builtin_bit_cast(__bf16, s);
}

// ---- CDNA5 async global->LDS copy (16B per lane), tracked by ASYNCcnt ------
DEV void cp_async_b128(void* lds, const void* g) {
#if __has_builtin(__builtin_amdgcn_global_load_async_to_lds_b128)
  __builtin_amdgcn_global_load_async_to_lds_b128(
      (gv4i_t*)g, (lv4i_t*)lds, 0, 0);
#else
  *reinterpret_cast<uint4*>(lds) = *reinterpret_cast<const uint4*>(g);
#endif
}

template <int N>
DEV void wait_async() {
#if __has_builtin(__builtin_amdgcn_s_wait_asynccnt)
  __builtin_amdgcn_s_wait_asynccnt(N);
#else
  asm volatile("s_wait_asynccnt %0" ::"i"(N) : "memory");
#endif
}

// A-matrix 16x32 (MxK) bf16 fragment, row-major source: A(m,k) = p[m*ld + k]
// lane m = lane&15, half = lane>>4; one contiguous 16-elem (32 B) run per lane (k = half*16 + e).
DEV v16bf load_a_rm(const __bf16* p, int ld) {
  int lane = threadIdx.x & 31;
  int m = lane & 15, half = lane >> 4;
  v16bf a;
#pragma unroll
  for (int j = 0; j < 8; ++j) {
    int k0 = (half << 4) + (j << 1);
    const __bf16* q = p + (size_t)m * ld + k0;
    a[2 * j]     = q[0];
    a[2 * j + 1] = q[1];
  }
  return a;
}

// B-matrix 32x16 (KxN) fragment from a COLUMN-MAJOR tile: B(k,n) = p[n*ldc + k]
// lane n = lane&15, half = lane>>4; fragment elem 2j+d has k = half*16 + 2j + d,
// i.e. one contiguous 16-element (32 B) run per lane -> ds_load_b128 x2.
DEV v16bf load_b_cm(const __bf16* p, int ldc) {
  int lane = threadIdx.x & 31;
  int n = lane & 15, half = lane >> 4;
  const __bf16* q = p + (size_t)n * ldc + half * 16;
  v16bf b;
#pragma unroll
  for (int e = 0; e < 16; ++e) b[e] = q[e];
  return b;
}

DEV v8f wmma_bf(v16bf a, v16bf b, v8f c) {
  return __builtin_amdgcn_wmma_f32_16x16x32_bf16(false, a, false, b,
                                                 (short)0, c, false, false);
}

// --------------------------- kernel 0: fp32 -> bf16 -------------------------
__global__ void k_convert(const float* __restrict__ src, __bf16* __restrict__ dst, int n) {
  int i = blockIdx.x * 256 + threadIdx.x;
  if (i < n) dst[i] = f2bf(src[i]);
}

// --------------------------- kernel 1: GroupNorm ----------------------------
// block per (b, group): 16 channels x 1024 spatial = 16384 elems.
// Output nbuf is T-MAJOR: n[b][t][c] (c contiguous) for column-major B tiles.
__global__ void k_groupnorm(const float* __restrict__ x,
                            const float* __restrict__ gw,
                            const float* __restrict__ gb,
                            __bf16* __restrict__ nbuf) {
  int blk = blockIdx.x;
  int b = blk >> 5, g = blk & 31;
  const float* base = x + ((size_t)(b * 512 + g * 16)) * 1024;
  __shared__ float ssum[256], ssq[256];
  float s = 0.f, ss = 0.f;
  for (int i = threadIdx.x; i < 16384; i += 256) {
    float v = base[i];
    s += v; ss += v * v;
  }
  ssum[threadIdx.x] = s; ssq[threadIdx.x] = ss;
  __syncthreads();
  for (int st = 128; st > 0; st >>= 1) {
    if (threadIdx.x < st) {
      ssum[threadIdx.x] += ssum[threadIdx.x + st];
      ssq[threadIdx.x]  += ssq[threadIdx.x + st];
    }
    __syncthreads();
  }
  float mu  = ssum[0] * (1.0f / 16384.0f);
  float var = ssq[0] * (1.0f / 16384.0f) - mu * mu;
  float rs  = rsqrtf(var + 1e-5f);
  __bf16* nb = nbuf + (size_t)b * 512 * 1024;
  // e = t*16 + chLocal ordering -> coalesced bf16 stores
  for (int e = threadIdx.x; e < 16384; e += 256) {
    int t = e >> 4, cl = e & 15;
    int ch = g * 16 + cl;
    float v = (base[(size_t)cl * 1024 + t] - mu) * rs * gw[ch] + gb[ch];
    nb[(size_t)t * 512 + ch] = f2bf(v);
  }
}

// --------------------------- kernel 2: QKV GEMM -----------------------------
// qkv[o,t] = sum_c W[o,c] * n[c,t] + bias[o]; n is t-major [t][c].
// WG: 8 waves, macro 32(o) x 256(t); per K-step stage A(32x32, row-major) and
// B slab (256 cols x 32 c, column-major) into LDS via async b128.
// Store routing per 16-row tile (tiles never cross q/k/v segment boundaries):
//   q: qT[bh][t][w]  * hd^-0.25   (t-major, feeds A of QK^T)
//   k: kT[bh][s][c]  * hd^-0.25   (s-major, feeds B of QK^T)
//   v: vC[bh][c][s]               (c-major, feeds B of P.V)
__global__ void k_qkv(const __bf16* __restrict__ W, const __bf16* __restrict__ N,
                      const float* __restrict__ bias,
                      __bf16* __restrict__ qT, __bf16* __restrict__ kT,
                      __bf16* __restrict__ vC) {
  __shared__ __bf16 Ash[32 * 32];    // 2 KB, row-major
  __shared__ __bf16 Bsh[256 * 32];   // 16 KB, column-major (col stride 32)
  int b   = blockIdx.z;
  int tid = threadIdx.x;
  int wv  = tid >> 5;
  int ot  = wv & 1, tc = wv >> 1;
  int ob  = blockIdx.x * 32;
  int tb  = blockIdx.y * 256;
  const __bf16* Nb = N + (size_t)b * 512 * 1024;  // [t][c], ld = 512

  v8f acc[4];
#pragma unroll
  for (int i = 0; i < 4; ++i) acc[i] = (v8f){0.f,0.f,0.f,0.f,0.f,0.f,0.f,0.f};

  for (int kc = 0; kc < 512; kc += 32) {
    __syncthreads();  // previous iteration's LDS reads done
    if (tid < 128) {  // A slab: 32 rows x 64B = 128 chunks
      int row = tid >> 2, h = tid & 3;
      cp_async_b128(Ash + row * 32 + h * 8,
                    W + (size_t)(ob + row) * 512 + kc + h * 8);
    }
#pragma unroll
    for (int q = 0; q < 4; ++q) {  // B slab: 256 cols x 64B = 1024 chunks
      int chunk = tid + 256 * q;
      int col = chunk >> 2, h = chunk & 3;
      cp_async_b128(Bsh + col * 32 + h * 8,
                    Nb + (size_t)(tb + col) * 512 + kc + h * 8);
    }
    wait_async<0>();
    __syncthreads();

    v16bf A = load_a_rm(Ash + ot * 16 * 32, 32);
#pragma unroll
    for (int i = 0; i < 4; ++i) {
      v16bf Bm = load_b_cm(Bsh + (tc * 64 + i * 16) * 32, 32);
      acc[i] = wmma_bf(A, Bm, acc[i]);
    }
  }

  const float s4 = 0.35355339059327373f;  // 64^-0.25
  int lane = threadIdx.x & 31;
  int n = lane & 15, half = lane >> 4;
  int o0 = ob + ot * 16, t0 = tb + tc * 64;
  // The whole 16-row tile lives in one q/k/v segment: decide once per wave.
  int head  = o0 / 192;                 // wave-uniform
  int w0    = o0 - head * 192;
  int seg   = w0 >> 6;                  // 0=q, 1=k, 2=v
  int wl0   = w0 & 63;
  int bh    = b * 8 + head;
  int mrow  = 8 * half;                 // lane's row offset within tile
  if (seg == 0) {
    __bf16* dst = qT + (size_t)bh * 1024 * 64;
#pragma unroll
    for (int i = 0; i < 4; ++i)
#pragma unroll
      for (int r = 0; r < 8; ++r) {
        int o = o0 + r + mrow, t = t0 + i * 16 + n;
        dst[(size_t)t * 64 + (wl0 + r + mrow)] = f2bf((acc[i][r] + bias[o]) * s4);
      }
  } else if (seg == 1) {
    __bf16* dst = kT + (size_t)bh * 1024 * 64;
#pragma unroll
    for (int i = 0; i < 4; ++i)
#pragma unroll
      for (int r = 0; r < 8; ++r) {
        int o = o0 + r + mrow, t = t0 + i * 16 + n;
        dst[(size_t)t * 64 + (wl0 + r + mrow)] = f2bf((acc[i][r] + bias[o]) * s4);
      }
  } else {
    __bf16* dst = vC + (size_t)bh * 64 * 1024;
#pragma unroll
    for (int i = 0; i < 4; ++i)
#pragma unroll
      for (int r = 0; r < 8; ++r) {
        int o = o0 + r + mrow, t = t0 + i * 16 + n;
        dst[(size_t)(wl0 + r + mrow) * 1024 + t] = f2bf(acc[i][r] + bias[o]);
      }
  }
}

// --------------------------- kernel 3: attention ----------------------------
// One WG per (bh, 32-query block). Dynamic LDS (240 KB < 320 KB/WGP):
//   S[32][1024] fp32 scores + per-wave double-buffered K/V bf16 stages +
//   Pb[32][1024] bf16 probabilities (written once in softmax, WMMA-A ready).
__global__ void k_attn(const __bf16* __restrict__ qT, const __bf16* __restrict__ kT,
                       const __bf16* __restrict__ vC, __bf16* __restrict__ obuf) {
  extern __shared__ char smem[];
  float*  S      = (float*)smem;                          // 131072 B
  __bf16* kstage = (__bf16*)(smem + 131072);              // 32768 B
  __bf16* vstage = (__bf16*)(smem + 131072 + 32768);      // 16384 B
  __bf16* Pb     = (__bf16*)(smem + 131072 + 32768 + 16384);  // 65536 B

  int bh = blockIdx.y;
  int t0 = blockIdx.x * 32;
  int wv = threadIdx.x >> 5;
  int lane = threadIdx.x & 31;
  int n = lane & 15, half = lane >> 4;

  // Q fragments: A(m,k) = q[c=k0+k][t0+m] from qT[bh][t][c] (row-major ld=64)
  v16bf aQ[2][2];
#pragma unroll
  for (int tt = 0; tt < 2; ++tt)
#pragma unroll
    for (int kk = 0; kk < 2; ++kk)
      aQ[tt][kk] = load_a_rm(qT + ((size_t)bh * 1024 + t0 + tt * 16) * 64 + kk * 32, 64);

  // ---- phase 1: S = (q*s).(k*s); wave wv owns s-tiles wv, wv+8, ...
  const __bf16* kbase = kT + (size_t)bh * 1024 * 64;  // [s][c]
  __bf16* myk = kstage + wv * 2048;  // two 1024-elem buffers (tile = 2 KB)

  // tile st: contiguous 2 KB at kbase + st*16*64; 128 chunks, 4 per lane
#pragma unroll
  for (int q = 0; q < 4; ++q) {
    int ch = lane + 32 * q;
    cp_async_b128(myk + ch * 8, kbase + (size_t)wv * 16 * 64 + ch * 8);
  }
  for (int idx = 0; idx < 8; ++idx) {
    int st = wv + idx * 8;
    if (idx < 7) {
      __bf16* nxt = myk + ((idx + 1) & 1) * 1024;
#pragma unroll
      for (int q = 0; q < 4; ++q) {
        int ch = lane + 32 * q;
        cp_async_b128(nxt + ch * 8, kbase + (size_t)(st + 8) * 16 * 64 + ch * 8);
      }
      wait_async<1>();
    } else {
      wait_async<0>();
    }
    const __bf16* kt = myk + (idx & 1) * 1024;  // [s_local][c], ld=64
    v16bf Bk0 = load_b_cm(kt + 0, 64);   // c rows 0..31
    v16bf Bk1 = load_b_cm(kt + 32, 64);  // c rows 32..63
#pragma unroll
    for (int tt = 0; tt < 2; ++tt) {
      v8f acc = (v8f){0.f,0.f,0.f,0.f,0.f,0.f,0.f,0.f};
      acc = wmma_bf(aQ[tt][0], Bk0, acc);
      acc = wmma_bf(aQ[tt][1], Bk1, acc);
#pragma unroll
      for (int r = 0; r < 8; ++r)
        S[(size_t)(tt * 16 + r + 8 * half) * 1024 + st * 16 + n] = acc[r];
    }
  }
  __syncthreads();

  // ---- phase 2: exact softmax; emit bf16 P panel (WMMA-A ready layout)
#pragma unroll
  for (int rr = 0; rr < 4; ++rr) {
    int row = wv * 4 + rr;
    float*  Sr = S  + (size_t)row * 1024;
    __bf16* Pr = Pb + (size_t)row * 1024;
    float mx = -3.0e38f;
    for (int j = lane; j < 1024; j += 32) mx = fmaxf(mx, Sr[j]);
#pragma unroll
    for (int d = 16; d > 0; d >>= 1) mx = fmaxf(mx, __shfl_xor(mx, d, 32));
    float sum = 0.f;
    for (int j = lane; j < 1024; j += 32) {
      float e = __expf(Sr[j] - mx);
      Sr[j] = e;
      sum += e;
    }
#pragma unroll
    for (int d = 16; d > 0; d >>= 1) sum += __shfl_xor(sum, d, 32);
    float inv = 1.0f / sum;
    for (int j = lane; j < 1024; j += 32) Pr[j] = f2bf(Sr[j] * inv);
  }
  __syncthreads();

  // ---- phase 3: O[t,c] = sum_s P[t,s]*v[c,s]; 8 tiles (2t x 4c), 1 per wave
  int tt = wv >> 2, ct = wv & 3;
  const __bf16* vbase = vC + (size_t)(bh * 64 + ct * 16) * 1024;  // [c][s]
  __bf16* myv = vstage + wv * 1024;  // two 512-elem buffers (16c x 32s, c-major)

  // stage V tile s0=0: 16 c rows x 64B = 64 chunks, 2 per lane
#pragma unroll
  for (int q = 0; q < 2; ++q) {
    int ch = lane + 32 * q;
    int cl = ch >> 2, h = ch & 3;
    cp_async_b128(myv + cl * 32 + h * 8, vbase + (size_t)cl * 1024 + h * 8);
  }
  v8f acc = (v8f){0.f,0.f,0.f,0.f,0.f,0.f,0.f,0.f};
  for (int i = 0; i < 32; ++i) {
    int s0 = i * 32;
    if (i < 31) {
      __bf16* nxt = myv + ((i + 1) & 1) * 512;
#pragma unroll
      for (int q = 0; q < 2; ++q) {
        int ch = lane + 32 * q;
        int cl = ch >> 2, h = ch & 3;
        cp_async_b128(nxt + cl * 32 + h * 8,
                      vbase + (size_t)cl * 1024 + s0 + 32 + h * 8);
      }
      wait_async<1>();
    } else {
      wait_async<0>();
    }
    v16bf A  = load_a_rm(Pb + (size_t)(tt * 16) * 1024 + s0, 1024);
    v16bf Bv = load_b_cm(myv + (i & 1) * 512, 32);
    acc = wmma_bf(A, Bv, acc);
  }
  // obuf is t-major: o[b][t][ch] (ch contiguous) -> coalesced 32B stores
  int b = bh >> 3, head = bh & 7;
  __bf16* ob = obuf + (size_t)b * 1024 * 512;
#pragma unroll
  for (int r = 0; r < 8; ++r) {
    int t  = t0 + tt * 16 + r + 8 * half;
    int ch = head * 64 + ct * 16 + n;
    ob[(size_t)t * 512 + ch] = f2bf(acc[r]);
  }
}

// --------------------------- kernel 4: out-proj + residual ------------------
// out[o,t] = sum_c Wout[o,c] * obuf[t][c] + out_b[o] + x[o,t]
__global__ void k_out(const __bf16* __restrict__ W, const __bf16* __restrict__ N,
                      const float* __restrict__ bias, const float* __restrict__ x,
                      float* __restrict__ out) {
  __shared__ __bf16 Ash[32 * 32];
  __shared__ __bf16 Bsh[256 * 32];
  int b   = blockIdx.z;
  int tid = threadIdx.x;
  int wv  = tid >> 5;
  int ot  = wv & 1, tc = wv >> 1;
  int ob  = blockIdx.x * 32;
  int tb  = blockIdx.y * 256;
  const __bf16* Nb = N + (size_t)b * 1024 * 512;  // [t][c], ld = 512

  v8f acc[4];
#pragma unroll
  for (int i = 0; i < 4; ++i) acc[i] = (v8f){0.f,0.f,0.f,0.f,0.f,0.f,0.f,0.f};

  for (int kc = 0; kc < 512; kc += 32) {
    __syncthreads();
    if (tid < 128) {
      int row = tid >> 2, h = tid & 3;
      cp_async_b128(Ash + row * 32 + h * 8,
                    W + (size_t)(ob + row) * 512 + kc + h * 8);
    }
#pragma unroll
    for (int q = 0; q < 4; ++q) {
      int chunk = tid + 256 * q;
      int col = chunk >> 2, h = chunk & 3;
      cp_async_b128(Bsh + col * 32 + h * 8,
                    Nb + (size_t)(tb + col) * 512 + kc + h * 8);
    }
    wait_async<0>();
    __syncthreads();

    v16bf A = load_a_rm(Ash + ot * 16 * 32, 32);
#pragma unroll
    for (int i = 0; i < 4; ++i) {
      v16bf Bm = load_b_cm(Bsh + (tc * 64 + i * 16) * 32, 32);
      acc[i] = wmma_bf(A, Bm, acc[i]);
    }
  }

  int lane = threadIdx.x & 31;
  int n = lane & 15, half = lane >> 4;
  int o0 = ob + ot * 16, t0 = tb + tc * 64;
#pragma unroll
  for (int i = 0; i < 4; ++i) {
#pragma unroll
    for (int r = 0; r < 8; ++r) {
      int o = o0 + r + 8 * half;
      int t = t0 + i * 16 + n;
      size_t idx = ((size_t)(b * 512 + o)) * 1024 + t;
      out[idx] = acc[i][r] + bias[o] + x[idx];
    }
  }
}

// ---------------------------------------------------------------------------
extern "C" void kernel_launch(void* const* d_in, const int* in_sizes, int n_in,
                              void* d_out, int out_size, void* d_ws, size_t ws_size,
                              hipStream_t stream) {
  const float* x     = (const float*)d_in[0];  // (8,512,32,32)
  const float* gn_w  = (const float*)d_in[1];  // (512)
  const float* gn_b  = (const float*)d_in[2];  // (512)
  const float* qkv_w = (const float*)d_in[3];  // (1536,512)
  const float* qkv_b = (const float*)d_in[4];  // (1536)
  const float* out_w = (const float*)d_in[5];  // (512,512)
  const float* out_b = (const float*)d_in[6];  // (512)
  float* out = (float*)d_out;

  // workspace layout (bf16 intermediates), total ~44 MB
  char* ws = (char*)d_ws;
  size_t off = 0;
  __bf16* wqkv = (__bf16*)(ws + off); off += (size_t)1536 * 512 * 2;
  __bf16* wout = (__bf16*)(ws + off); off += (size_t)512 * 512 * 2;
  __bf16* nbuf = (__bf16*)(ws + off); off += (size_t)8 * 1024 * 512 * 2;   // [b][t][c]
  __bf16* qT   = (__bf16*)(ws + off); off += (size_t)64 * 1024 * 64 * 2;   // [bh][t][c]
  __bf16* kT   = (__bf16*)(ws + off); off += (size_t)64 * 1024 * 64 * 2;   // [bh][s][c]
  __bf16* vC   = (__bf16*)(ws + off); off += (size_t)64 * 64 * 1024 * 2;   // [bh][c][s]
  __bf16* obuf = (__bf16*)(ws + off); off += (size_t)8 * 1024 * 512 * 2;   // [b][t][c]
  (void)ws_size; (void)in_sizes; (void)n_in; (void)out_size;

  // 0) weights -> bf16
  k_convert<<<(1536 * 512 + 255) / 256, 256, 0, stream>>>(qkv_w, wqkv, 1536 * 512);
  k_convert<<<(512 * 512 + 255) / 256, 256, 0, stream>>>(out_w, wout, 512 * 512);

  // 1) group norm -> bf16 activations (t-major)
  k_groupnorm<<<8 * 32, 256, 0, stream>>>(x, gn_w, gn_b, nbuf);

  // 2) QKV projection (WMMA + async LDS staging)
  k_qkv<<<dim3(48, 4, 8), 256, 0, stream>>>(wqkv, nbuf, qkv_b, qT, kT, vC);

  // 3) attention: 64 head-batches x 32 query-blocks
  //    LDS: 128 KB scores + 32 KB K-stage + 16 KB V-stage + 64 KB P = 240 KB
  k_attn<<<dim3(32, 64), 256, 131072 + 32768 + 16384 + 65536, stream>>>(qT, kT, vC, obuf);

  // 4) output projection + bias + residual (fp32 out)
  k_out<<<dim3(16, 4, 8), 256, 0, stream>>>(wout, obuf, out_b, x, out);
}
